// FlatQuantMLA_52527450030274
// MI455X (gfx1250) — matrix operand order repaired and checked by reading, not compiled, not run on hardware
//
#include <hip/hip_runtime.h>
#include <hip/hip_bf16.h>

typedef __attribute__((ext_vector_type(16))) __bf16 v16bf;
typedef __attribute__((ext_vector_type(8)))  float  v8f;

#define B_    2
#define S_    2048
#define DIM_  2048
#define H_    16
#define C_    512
#define NOPE_ 128
#define ROPE_ 64
#define V_    128
#define QK_   192
#define HQK_  (H_*QK_)   /* 3072 */
#define KE_   (C_+ROPE_) /* 576  */
#define SCALE_ 0.07216878364870322f /* 192^-0.5 */

union U8bf { uint4 u; __bf16 h[8]; };

__device__ __forceinline__ v8f vzero8() {
  v8f r = {0.f,0.f,0.f,0.f,0.f,0.f,0.f,0.f};
  return r;
}

// 16 bf16 fragment from two 8-float chunks (f32 source, convert inline)
__device__ __forceinline__ v16bf cvt16_f32(const float* p0, const float* p1) {
  float4 a0 = ((const float4*)p0)[0];
  float4 a1 = ((const float4*)p0)[1];
  float4 b0 = ((const float4*)p1)[0];
  float4 b1 = ((const float4*)p1)[1];
  v16bf r;
  r[0]=(__bf16)a0.x;  r[1]=(__bf16)a0.y;  r[2]=(__bf16)a0.z;  r[3]=(__bf16)a0.w;
  r[4]=(__bf16)a1.x;  r[5]=(__bf16)a1.y;  r[6]=(__bf16)a1.z;  r[7]=(__bf16)a1.w;
  r[8]=(__bf16)b0.x;  r[9]=(__bf16)b0.y;  r[10]=(__bf16)b0.z; r[11]=(__bf16)b0.w;
  r[12]=(__bf16)b1.x; r[13]=(__bf16)b1.y; r[14]=(__bf16)b1.z; r[15]=(__bf16)b1.w;
  return r;
}

// 16 bf16 fragment from two 16-byte chunks (bf16 source, global or LDS)
__device__ __forceinline__ v16bf ld16_bf16(const __bf16* p0, const __bf16* p1) {
  U8bf a, b;
  a.u = *(const uint4*)p0;
  b.u = *(const uint4*)p1;
  v16bf r;
#pragma unroll
  for (int i = 0; i < 8; ++i) { r[i] = a.h[i]; r[8+i] = b.h[i]; }
  return r;
}

__device__ __forceinline__ v8f wmma_bf16(v16bf a, v16bf b, v8f c) {
  return __builtin_amdgcn_wmma_f32_16x16x32_bf16(false, a, false, b, (short)0, c,
                                                 false, false);
}

// ---------------------------------------------------------------------------
// one-shot f32 -> bf16 staging (vectorized, bandwidth-bound)
// ---------------------------------------------------------------------------
__global__ void __launch_bounds__(256)
cvt_bf16_k(const float* __restrict__ in, __bf16* __restrict__ out)
{
  size_t i = ((size_t)blockIdx.x * 256 + threadIdx.x) * 4;
  float4 v = *(const float4*)(in + i);
  __bf16* o = out + i;
  o[0] = (__bf16)v.x; o[1] = (__bf16)v.y; o[2] = (__bf16)v.z; o[3] = (__bf16)v.w;
}

// ---------------------------------------------------------------------------
// Generic NT GEMM, bf16 x bf16: C[M,N] = A[M,K] * W[N,K]^T
// 2x M-register blocking: wave computes 32x16, reusing the B fragment.
// grid: (N/64, M/32), block 128 (4 waves)
// ---------------------------------------------------------------------------
template <typename OutT>
__global__ void __launch_bounds__(128)
gemm_nt_bf16(const __bf16* __restrict__ A, int lda,
             const __bf16* __restrict__ W, int ldb,
             OutT* __restrict__ Cout, int ldc, int K)
{
  const int lane = threadIdx.x & 31, wid = threadIdx.x >> 5;
  const int lr = lane & 15, h8 = (lane >> 4) * 8, h16 = (lane >> 4) * 16;
  const int row0 = blockIdx.y * 32;
  const int col0 = blockIdx.x * 64 + wid * 16;
  const __bf16* arow0 = A + (size_t)(row0 + lr) * lda;
  const __bf16* arow1 = A + (size_t)(row0 + 16 + lr) * lda;
  const __bf16* brow  = W + (size_t)(col0 + lr) * ldb;
  v8f acc0 = vzero8(), acc1 = vzero8();
#pragma unroll 4
  for (int k = 0; k < K; k += 32) {
    v16bf b  = ld16_bf16(brow  + k + h8,  brow  + k + 16 + h8);
    v16bf a0 = ld16_bf16(arow0 + k + h8,  arow0 + k + 16 + h8);
    v16bf a1 = ld16_bf16(arow1 + k + h8,  arow1 + k + 16 + h8);
    acc0 = wmma_bf16(a0, b, acc0);
    acc1 = wmma_bf16(a1, b, acc1);
  }
#pragma unroll
  for (int i = 0; i < 8; ++i) {
    int m = row0 + i + 8 * (lane >> 4);
    Cout[(size_t)m * ldc + col0 + lr]        = (OutT)acc0[i];
    Cout[(size_t)(m + 16) * ldc + col0 + lr] = (OutT)acc1[i];
  }
}

// ---------------------------------------------------------------------------
// kv_eff row: rmsnorm(kv)*w (512) ++ rope(k_pe) (64), bf16. grid B*S, block 256
// ---------------------------------------------------------------------------
__global__ void __launch_bounds__(256)
kv_post(const float* __restrict__ kvpe, const float* __restrict__ w,
        const float* __restrict__ cosT, const float* __restrict__ sinT,
        __bf16* __restrict__ kv_eff)
{
  __shared__ float red[256];
  const int r = blockIdx.x, t = threadIdx.x;
  const int s = r & (S_ - 1);
  const float* row = kvpe + (size_t)r * KE_;
  float v0 = row[t], v1 = row[t + 256];
  red[t] = v0 * v0 + v1 * v1;
  __syncthreads();
  for (int off = 128; off > 0; off >>= 1) {
    if (t < off) red[t] += red[t + off];
    __syncthreads();
  }
  float rms = rsqrtf(red[0] * (1.0f / (float)C_) + 1e-6f);
  __bf16* out = kv_eff + (size_t)r * KE_;
  out[t]       = (__bf16)(v0 * rms * w[t]);
  out[t + 256] = (__bf16)(v1 * rms * w[t + 256]);
  if (t < 32) {
    float a = row[C_ + 2*t], b = row[C_ + 2*t + 1];
    float c = cosT[s * 32 + t], sn = sinT[s * 32 + t];
    out[C_ + 2*t]     = (__bf16)(a * c - b * sn);
    out[C_ + 2*t + 1] = (__bf16)(a * sn + b * c);
  }
}

// wb1t[h][c][d] = wkv_b[h*256 + d][c]   (bf16, NT-friendly for q_abs)
__global__ void __launch_bounds__(256)
build_wb1t(const float* __restrict__ wkv_b, __bf16* __restrict__ wb1t)
{
  int idx = blockIdx.x * 256 + threadIdx.x;       // h:4 | c:9 | d:7
  int h = idx >> 16, rem = idx & 65535;
  int c = rem >> 7, d = rem & 127;
  wb1t[idx] = (__bf16)wkv_b[(size_t)(h * 256 + d) * C_ + c];
}

// ---------------------------------------------------------------------------
// q_abs = q_nope @ wb1, scaled, into q_eff[:, :512].  (bf16 x bf16)
// grid (C/64, S/16, B*H), block 128
// ---------------------------------------------------------------------------
__global__ void __launch_bounds__(128)
qabs_k(const __bf16* __restrict__ q_bf, const __bf16* __restrict__ wb1t,
       __bf16* __restrict__ q_eff)
{
  const int lane = threadIdx.x & 31, wid = threadIdx.x >> 5;
  const int lr = lane & 15, h8 = (lane >> 4) * 8, h16 = (lane >> 4) * 16;
  const int bh = blockIdx.z, b = bh >> 4, h = bh & 15;
  const int row0 = blockIdx.y * 16;
  const int col0 = blockIdx.x * 64 + wid * 16;
  const __bf16* arow = q_bf + (size_t)(b * S_ + row0 + lr) * HQK_ + h * QK_;
  const __bf16* brow = wb1t + (size_t)(h * C_ + col0 + lr) * NOPE_;
  v8f acc = vzero8();
#pragma unroll
  for (int k = 0; k < NOPE_; k += 32) {
    v16bf a  = ld16_bf16(arow + k + h8,  arow + k + 16 + h8);
    v16bf bb = ld16_bf16(brow + k + h8,  brow + k + 16 + h8);
    acc = wmma_bf16(a, bb, acc);
  }
  __bf16* orow = q_eff + (size_t)bh * S_ * KE_;
#pragma unroll
  for (int i = 0; i < 8; ++i) {
    int m = row0 + i + 8 * (lane >> 4);
    orow[(size_t)m * KE_ + col0 + lr] = (__bf16)(acc[i] * SCALE_);
  }
}

// rope(q_pe)*SCALE into q_eff[:, 512:576]
__global__ void __launch_bounds__(256)
qpe_k(const __bf16* __restrict__ q_bf,
      const float* __restrict__ cosT, const float* __restrict__ sinT,
      __bf16* __restrict__ q_eff)
{
  int idx = blockIdx.x * 256 + threadIdx.x;   // i:5 | h:4 | s:11 | b:1
  int i = idx & 31; int rest = idx >> 5;
  int h = rest & (H_ - 1); rest >>= 4;
  int s = rest & (S_ - 1); int b = rest >> 11;
  const __bf16* qrow = q_bf + (size_t)(b * S_ + s) * HQK_ + h * QK_ + NOPE_;
  float a = (float)qrow[2*i], bb = (float)qrow[2*i + 1];
  float c = cosT[s * 32 + i], sn = sinT[s * 32 + i];
  __bf16* out = q_eff + ((size_t)(b * H_ + h) * S_ + s) * KE_ + C_;
  out[2*i]     = (__bf16)((a * c - bb * sn) * SCALE_);
  out[2*i + 1] = (__bf16)((a * sn + bb * c) * SCALE_);
}

// ---------------------------------------------------------------------------
// Flash attention. grid (S/16, B*H), block 64 (2 waves; wave owns 256 of C=512)
// ---------------------------------------------------------------------------
__global__ void __launch_bounds__(64)
attn_k(const __bf16* __restrict__ q_eff, const __bf16* __restrict__ kv_eff,
       __bf16* __restrict__ o_c)
{
  __shared__ __bf16 VT[C_][40];       // V tile transposed: [c][key], padded
  __shared__ __bf16 Pb[2][16][40];    // per-wave probabilities, A-relayout

  const int tid  = threadIdx.x;
  const int lane = tid & 31, wid = tid >> 5;
  const int lr = lane & 15, hh = lane >> 4;
  const int h8 = hh * 8, h16 = hh * 16;
  const int bh = blockIdx.y, b = bh >> 4;
  const int qbase = blockIdx.x * 16;
  const __bf16* qe  = q_eff + ((size_t)bh * S_ + qbase + lr) * KE_;
  const __bf16* kvb = kv_eff + (size_t)b * S_ * KE_;
  const int cq = wid * 256;

  float m_st[8], l_st[8];
  v8f oacc[16];
#pragma unroll
  for (int i = 0; i < 8; ++i) { m_st[i] = -1e30f; l_st[i] = 0.f; }
#pragma unroll
  for (int t = 0; t < 16; ++t) oacc[t] = vzero8();

  const int nkt = ((qbase + 15) >> 5) + 1;

  for (int kt = 0; kt < nkt; ++kt) {
    const int kb = kt * 32;
    __syncthreads();                       // protect VT/Pb from prior iter
    {   // stage V^T (32 keys x 512 latent) into LDS, 2 threads per key
      int key = tid >> 1;
      int c0  = (tid & 1) * 256;
      const __bf16* src = kvb + (size_t)(kb + key) * KE_ + c0;
#pragma unroll
      for (int cc = 0; cc < 256; cc += 8) {
        U8bf u; u.u = *(const uint4*)(src + cc);
#pragma unroll
        for (int j = 0; j < 8; ++j) VT[c0 + cc + j][key] = u.h[j];
      }
    }
    __syncthreads();

    // scores: 16 q-rows x 32 keys, K=576
    v8f s0 = vzero8(), s1 = vzero8();
    const __bf16* k0r = kvb + (size_t)(kb + lr) * KE_;
    const __bf16* k1r = kvb + (size_t)(kb + 16 + lr) * KE_;
#pragma unroll
    for (int k = 0; k < KE_; k += 32) {
      v16bf a  = ld16_bf16(qe + k + h16, qe + k + h16 + 8);
      v16bf b0 = ld16_bf16(k0r + k + h16, k0r + k + h16 + 8);
      s0 = wmma_bf16(a, b0, s0);
      v16bf b1 = ld16_bf16(k1r + k + h16, k1r + k + h16 + 8);
      s1 = wmma_bf16(a, b1, s1);
    }

    // causal mask + online softmax (rows live per 16-lane half-group)
#pragma unroll
    for (int i = 0; i < 8; ++i) {
      int m = i + 8 * hh;
      int qrow = qbase + m;
      float v0 = (kb + lr      <= qrow) ? s0[i] : -1e30f;
      float v1 = (kb + 16 + lr <= qrow) ? s1[i] : -1e30f;
      float t = fmaxf(v0, v1);
#pragma unroll
      for (int off = 1; off < 16; off <<= 1)
        t = fmaxf(t, __shfl_xor(t, off, 32));
      float mnew = fmaxf(m_st[i], t);
      float sf = __expf(m_st[i] - mnew);
      float p0 = __expf(v0 - mnew);
      float p1 = __expf(v1 - mnew);
      float rs = p0 + p1;
#pragma unroll
      for (int off = 1; off < 16; off <<= 1)
        rs += __shfl_xor(rs, off, 32);
      l_st[i] = l_st[i] * sf + rs;
      m_st[i] = mnew;
#pragma unroll
      for (int t2 = 0; t2 < 16; ++t2) oacc[t2][i] = oacc[t2][i] * sf;
      Pb[wid][m][lr]      = (__bf16)p0;
      Pb[wid][m][16 + lr] = (__bf16)p1;
    }
    __syncthreads();

    if (kt + 1 < nkt)   // L2 prefetch of next key tile
      __builtin_prefetch(kvb + (size_t)(kb + 32 + lane) * KE_, 0, 1);

    // PV: oacc(16x256) += P(16x32) * V(32x256)
    v16bf pa = ld16_bf16(&Pb[wid][lr][h16], &Pb[wid][lr][h16 + 8]);
#pragma unroll
    for (int t2 = 0; t2 < 16; ++t2) {
      const __bf16* vp = &VT[cq + t2 * 16 + lr][h16];
      v16bf bv = ld16_bf16(vp, vp + 8);
      oacc[t2] = wmma_bf16(pa, bv, oacc[t2]);
    }
  }

  // normalize and store bf16
  __bf16* orow = o_c + ((size_t)bh * S_ + qbase) * C_;
#pragma unroll
  for (int t2 = 0; t2 < 16; ++t2) {
#pragma unroll
    for (int i = 0; i < 8; ++i) {
      int m = i + 8 * hh;
      float val = oacc[t2][i] / l_st[i];
      orow[(size_t)m * C_ + cq + t2 * 16 + lr] = (__bf16)val;
    }
  }
}

// ---------------------------------------------------------------------------
// per-head output projection: o_head[b,s,h*V+v] = o_c[b,h,s,:] . wb2[h][v][:]
// grid (V/64, S/16, B*H), block 128  (bf16 x bf16)
// ---------------------------------------------------------------------------
__global__ void __launch_bounds__(128)
oproj_k(const __bf16* __restrict__ o_c, const __bf16* __restrict__ wkvb_bf,
        __bf16* __restrict__ o_head)
{
  const int lane = threadIdx.x & 31, wid = threadIdx.x >> 5;
  const int lr = lane & 15, h8 = (lane >> 4) * 8, h16 = (lane >> 4) * 16;
  const int bh = blockIdx.z, b = bh >> 4, h = bh & 15;
  const int row0 = blockIdx.y * 16;
  const int col0 = blockIdx.x * 64 + wid * 16;
  const __bf16* arow = o_c + ((size_t)bh * S_ + row0 + lr) * C_;
  const __bf16* brow = wkvb_bf + (size_t)(h * 256 + NOPE_ + col0 + lr) * C_;
  v8f acc = vzero8();
#pragma unroll 4
  for (int k = 0; k < C_; k += 32) {
    v16bf a  = ld16_bf16(arow + k + h8,  arow + k + 16 + h8);
    v16bf bb = ld16_bf16(brow + k + h8,  brow + k + 16 + h8);
    acc = wmma_bf16(a, bb, acc);
  }
#pragma unroll
  for (int i = 0; i < 8; ++i) {
    int m = row0 + i + 8 * (lane >> 4);
    o_head[(size_t)(b * S_ + m) * (H_ * V_) + h * V_ + col0 + lr] = (__bf16)acc[i];
  }
}

// ---------------------------------------------------------------------------
extern "C" void kernel_launch(void* const* d_in, const int* in_sizes, int n_in,
                              void* d_out, int out_size, void* d_ws, size_t ws_size,
                              hipStream_t stream)
{
  (void)in_sizes; (void)n_in; (void)out_size; (void)ws_size;
  const float* x        = (const float*)d_in[0];
  const float* wq       = (const float*)d_in[1];
  const float* wkv_a    = (const float*)d_in[2];
  const float* wkv_b    = (const float*)d_in[3];
  const float* wo       = (const float*)d_in[4];
  const float* kvnw     = (const float*)d_in[5];
  const float* cosT     = (const float*)d_in[6];
  const float* sinT     = (const float*)d_in[7];
  // d_in[8]=mask (causal, rebuilt in-kernel), d_in[9]=start_pos (0)
  float* outp = (float*)d_out;

  char* w = (char*)d_ws;
  size_t off = 0;
  auto take = [&](size_t bytes) -> void* {
    void* p = w + off;
    off += (bytes + 255) & ~(size_t)255;
    return p;
  };
  const size_t BS = (size_t)B_ * S_;
  // bf16 staging of all f32 operands (one pass each)
  __bf16* x_bf    = (__bf16*)take(BS * DIM_ * sizeof(__bf16));
  __bf16* wq_bf   = (__bf16*)take((size_t)HQK_ * DIM_ * sizeof(__bf16));
  __bf16* wkva_bf = (__bf16*)take((size_t)KE_ * DIM_ * sizeof(__bf16));
  __bf16* wo_bf   = (__bf16*)take((size_t)DIM_ * DIM_ * sizeof(__bf16));
  __bf16* wkvb_bf = (__bf16*)take((size_t)(H_ * 256) * C_ * sizeof(__bf16));
  // intermediates
  float*  kvpe   = (float*) take(BS * KE_ * sizeof(float));
  __bf16* q_bf   = (__bf16*)take(BS * HQK_ * sizeof(__bf16));
  __bf16* kv_eff = (__bf16*)take(BS * KE_ * sizeof(__bf16));
  __bf16* q_eff  = (__bf16*)take((size_t)B_ * H_ * S_ * KE_ * sizeof(__bf16));
  __bf16* wb1t   = (__bf16*)take((size_t)H_ * C_ * NOPE_ * sizeof(__bf16));
  __bf16* o_c    = (__bf16*)take((size_t)B_ * H_ * S_ * C_ * sizeof(__bf16));
  __bf16* o_head = (__bf16*)take(BS * (size_t)(H_ * V_) * sizeof(__bf16));

  // 0) stage operands to bf16
  cvt_bf16_k<<<dim3((BS * DIM_) / 1024), 256, 0, stream>>>(x, x_bf);
  cvt_bf16_k<<<dim3(((size_t)HQK_ * DIM_) / 1024), 256, 0, stream>>>(wq, wq_bf);
  cvt_bf16_k<<<dim3(((size_t)KE_ * DIM_) / 1024), 256, 0, stream>>>(wkv_a, wkva_bf);
  cvt_bf16_k<<<dim3(((size_t)DIM_ * DIM_) / 1024), 256, 0, stream>>>(wo, wo_bf);
  cvt_bf16_k<<<dim3(((size_t)H_ * 256 * C_) / 1024), 256, 0, stream>>>(wkv_b, wkvb_bf);

  // 1) kvpe = x @ wkv_a^T   (M=4096, N=576, K=2048), f32 out for rmsnorm
  gemm_nt_bf16<float><<<dim3(KE_ / 64, BS / 32), 128, 0, stream>>>(
      x_bf, DIM_, wkva_bf, DIM_, kvpe, KE_, DIM_);
  // 2) q = x @ wq^T         (M=4096, N=3072, K=2048), bf16 out
  gemm_nt_bf16<__bf16><<<dim3(HQK_ / 64, BS / 32), 128, 0, stream>>>(
      x_bf, DIM_, wq_bf, DIM_, q_bf, HQK_, DIM_);
  // 3) kv_eff = [rmsnorm(kv)*w, rope(k_pe)]
  kv_post<<<dim3(BS), 256, 0, stream>>>(kvpe, kvnw, cosT, sinT, kv_eff);
  // 4) wb1 transpose (bf16)
  build_wb1t<<<dim3((H_ * C_ * NOPE_) / 256), 256, 0, stream>>>(wkv_b, wb1t);
  // 5) q_eff[:, :512] = SCALE * q_nope @ wb1
  qabs_k<<<dim3(C_ / 64, S_ / 16, B_ * H_), 128, 0, stream>>>(q_bf, wb1t, q_eff);
  // 6) q_eff[:, 512:] = SCALE * rope(q_pe)
  qpe_k<<<dim3((B_ * S_ * H_ * 32) / 256), 256, 0, stream>>>(q_bf, cosT, sinT, q_eff);
  // 7) flash attention over latent
  attn_k<<<dim3(S_ / 16, B_ * H_), 64, 0, stream>>>(q_eff, kv_eff, o_c);
  // 8) per-head V projection
  oproj_k<<<dim3(V_ / 64, S_ / 16, B_ * H_), 128, 0, stream>>>(o_c, wkvb_bf, o_head);
  // 9) final output projection: out = o_head @ wo^T (f32 out)
  gemm_nt_bf16<float><<<dim3(DIM_ / 64, BS / 32), 128, 0, stream>>>(
      o_head, DIM_, wo_bf, DIM_, outp, DIM_, DIM_);
}
